// MixedMoE_90640989815288
// MI455X (gfx1250) — matrix operand order repaired and checked by reading, not compiled, not run on hardware
//
#include <hip/hip_runtime.h>
#include <hip/hip_bf16.h>

// ---------------------------------------------------------------------------
// MixedMoE for MI455X (gfx1250): sparse top-2 routing + bf16 WMMA GEMMs
// 64x64 tiles, double-buffered LDS, TDM (tensor_load_to_lds) for bf16 A tiles
// ---------------------------------------------------------------------------

typedef __bf16 bf16_t;
typedef __attribute__((ext_vector_type(16))) __bf16 v16bf;
typedef __attribute__((ext_vector_type(8)))  __bf16 v8bf;
typedef __attribute__((ext_vector_type(2)))  __bf16 v2bf;
typedef __attribute__((ext_vector_type(8)))  float  v8f;
typedef __attribute__((ext_vector_type(4)))  unsigned int v4u;
typedef __attribute__((ext_vector_type(8)))  int v8i;
typedef __attribute__((ext_vector_type(4)))  int v4i;

#define T_TOK  4096      // B*S tokens
#define DIM    1024
#define HDIM   1024
#define NEXP   16
#define NIN    2
#define NSHH   2048      // NSH * H
#define BM     64
#define BN     64
#define BK     32
#define LROW   56        // LDS row stride (32 data + 24 pad): n*28 mod 64 distinct

// ----- WMMA / fragment helpers ---------------------------------------------

__device__ __forceinline__ v8f wmma_bf16(v16bf a, v16bf b, v8f c) {
    return __builtin_amdgcn_wmma_f32_16x16x32_bf16(false, a, false, b,
                                                   (short)0, c, false, false);
}

// A fragment (16x32 bf16, row-major, stride LROW):
// lane m=l&15, h=l>>4; V0..3 = K[8h..8h+7], V4..7 = K[16+8h..16+8h+7]
__device__ __forceinline__ v16bf fragA(const bf16_t* base, int lane) {
    int r = lane & 15, h = lane >> 4;
    const bf16_t* p = base + r * LROW + 8 * h;
    union { v16bf v; v8bf q[2]; } u;
    u.q[0] = *(const v8bf*)(p);
    u.q[1] = *(const v8bf*)(p + 16);
    return u.v;
}

// B fragment (32x16 bf16, staged K-contiguous per N row, stride LROW):
// lane n=l&15, h=l>>4; V0..3 = K[8h..8h+7], V4..7 = K[16+8h..16+8h+7]
__device__ __forceinline__ v16bf fragB(const bf16_t* base, int lane) {
    int n = lane & 15, h = lane >> 4;
    const bf16_t* p = base + n * LROW + 8 * h;
    union { v16bf v; v8bf q[2]; } u;
    u.q[0] = *(const v8bf*)(p);
    u.q[1] = *(const v8bf*)(p + 16);
    return u.v;
}

__device__ __forceinline__ float silu_f(float x) {
    return x / (1.0f + __expf(-x));
}

// ----- TDM: DMA a 64x32 bf16 tile (row stride `stride_elems`) into LDS ------
// LDS pad: every 16 DWORDs insert 12 DWORDs => row stride 56 bf16 == LROW.
__device__ __forceinline__ void tdm_load_tile64x32(unsigned lds_addr,
                                                   const bf16_t* gsrc,
                                                   unsigned stride_elems) {
    unsigned long long ga = (unsigned long long)(uintptr_t)gsrc;
    v4u g0;
    g0[0] = 1u;                                           // count=1 (valid), load
    g0[1] = lds_addr;                                     // LDS byte address
    g0[2] = (unsigned)(ga & 0xffffffffu);                 // global_addr[31:0]
    g0[3] = (unsigned)((ga >> 32) & 0x01ffffffu) | (2u << 30);  // addr[56:32], type=2
    v8i g1;
    g1[0] = (1 << 16)        // data_size = 2 bytes
          | (1 << 20)        // pad_enable
          | (3 << 22)        // pad_interval: 16 DWORDs
          | (11 << 25);      // pad_amount: 12 DWORDs  -> 56 bf16 row stride
    g1[1] = (32 << 16);      // tensor_dim0[15:0] = 32
    g1[2] = (64 << 16);      // tensor_dim0 hi = 0 ; tensor_dim1[15:0] = 64
    g1[3] = (32 << 16);      // tensor_dim1 hi = 0 ; tile_dim0 = 32
    g1[4] = 64;              // tile_dim1 = 64 ; tile_dim2 = 0
    g1[5] = (int)stride_elems;  // tensor_dim0_stride[31:0] (data_size units)
    g1[6] = 0;
    g1[7] = 0;
    v4i z4 = {0, 0, 0, 0};
#if __clang_major__ >= 23
    v8i z8 = {0, 0, 0, 0, 0, 0, 0, 0};
    __builtin_amdgcn_tensor_load_to_lds(g0, g1, z4, z4, z8, 0);
#else
    __builtin_amdgcn_tensor_load_to_lds(g0, g1, z4, z4, 0);
#endif
}

// Stage B tile: 32(K) x 64(N) f32 -> bf16 transposed [n][k] in LDS.
// Thread owns a K-pair x 4 N: cvt_pk packs (k,k+1) -> ds_store_b32 (pair-merged).
__device__ __forceinline__ void stage_b_tile(const float* __restrict__ W, int ldw,
                                             int k0, int nb, int tid,
                                             bf16_t* __restrict__ dst) {
    int kp = tid >> 4, n4 = (tid & 15) * 4;
    const float* p0 = W + (size_t)(k0 + 2 * kp) * ldw + nb + n4;
    __builtin_prefetch(p0 + (size_t)2 * BK * ldw, 0, 3);   // global_prefetch_b8, k0+64
    float4 r0 = *(const float4*)(p0);
    float4 r1 = *(const float4*)(p0 + ldw);
    float a0[4] = {r0.x, r0.y, r0.z, r0.w};
    float a1[4] = {r1.x, r1.y, r1.z, r1.w};
#pragma unroll
    for (int j = 0; j < 4; ++j) {
        v2bf pk; pk[0] = (bf16_t)a0[j]; pk[1] = (bf16_t)a1[j];
        *(v2bf*)(dst + (n4 + j) * LROW + 2 * kp) = pk;
    }
}

// Stage A tile: 64 x 32 f32 -> bf16 row-major; one ds_store_b128 per thread.
__device__ __forceinline__ void stage_a_f32(const float* __restrict__ src, int ldk,
                                            int k0, int tid,
                                            bf16_t* __restrict__ dst) {
    int row = tid >> 2, c8 = (tid & 3) * 8;
    const float* p = src + (size_t)row * ldk + k0 + c8;
    float4 r0 = *(const float4*)(p);
    float4 r1 = *(const float4*)(p + 4);
    float a[8] = {r0.x, r0.y, r0.z, r0.w, r1.x, r1.y, r1.z, r1.w};
    v8bf pk;
#pragma unroll
    for (int j = 0; j < 8; ++j) pk[j] = (bf16_t)a[j];
    *(v8bf*)(dst + row * LROW + c8) = pk;
}

// ----- Kernel 0: zero per-expert counters -----------------------------------

__global__ void moe_init_kernel(int* counts) {
    if (threadIdx.x < NEXP) counts[threadIdx.x] = 0;
}

// ----- Kernel 1: gating (softmax + top-2 + compacted routing lists) ---------

__global__ __launch_bounds__(256)
void moe_gate_kernel(const float* __restrict__ emb, const float* __restrict__ gw,
                     int* __restrict__ counts, int* __restrict__ tokL,
                     float* __restrict__ wL) {
    const int t   = blockIdx.x;
    const int tid = threadIdx.x;
    float acc[NEXP];
#pragma unroll
    for (int e = 0; e < NEXP; ++e) acc[e] = 0.0f;

    for (int d = tid; d < DIM; d += 256) {
        float ev = emb[t * DIM + d];
#pragma unroll
        for (int e = 0; e < NEXP; ++e) acc[e] += ev * gw[e * DIM + d];
    }

    __shared__ float red[NEXP * 256];   // [e][tid]: conflict-free
#pragma unroll
    for (int e = 0; e < NEXP; ++e) red[e * 256 + tid] = acc[e];
    __syncthreads();
    for (int s = 128; s >= 1; s >>= 1) {
        if (tid < s) {
#pragma unroll
            for (int e = 0; e < NEXP; ++e)
                red[e * 256 + tid] += red[e * 256 + tid + s];
        }
        __syncthreads();
    }

    if (tid == 0) {
        float mx = red[0];
#pragma unroll
        for (int e = 1; e < NEXP; ++e) mx = fmaxf(mx, red[e * 256]);
        float p[NEXP], sum = 0.0f;
#pragma unroll
        for (int e = 0; e < NEXP; ++e) { p[e] = __expf(red[e * 256] - mx); sum += p[e]; }
        float inv = 1.0f / sum;
        int   i1 = 0; float v1 = -1.0f;
#pragma unroll
        for (int e = 0; e < NEXP; ++e) if (p[e] > v1) { v1 = p[e]; i1 = e; }
        int   i2 = -1; float v2 = -1.0f;
#pragma unroll
        for (int e = 0; e < NEXP; ++e)
            if (e != i1 && p[e] > v2) { v2 = p[e]; i2 = e; }

        int pos = atomicAdd(&counts[i1], 1);
        tokL[i1 * T_TOK + pos] = t; wL[i1 * T_TOK + pos] = v1 * inv;
        pos = atomicAdd(&counts[i2], 1);
        tokL[i2 * T_TOK + pos] = t; wL[i2 * T_TOK + pos] = v2 * inv;
    }
}

// ----- Kernel 2: exclusive scan of expert counts ----------------------------

__global__ void moe_scan_kernel(const int* __restrict__ counts, int* __restrict__ offs) {
    if (threadIdx.x == 0) {
        int s = 0;
#pragma unroll
        for (int e = 0; e < NEXP; ++e) { offs[e] = s; s += counts[e]; }
    }
}

// ----- Kernel 3: shared MLP stage 1: hs = silu(emb @ sW1 + sB1) (bf16) ------
// grid (T/64, NSHH/64), block 256; 8 waves = 4 M-rows x 2 N-pairs

__global__ __launch_bounds__(256)
void moe_shared1_kernel(const float* __restrict__ emb, const float* __restrict__ sW1,
                        const float* __restrict__ sB1, bf16_t* __restrict__ hs) {
    __shared__ __attribute__((aligned(16))) bf16_t ldsA[2][BM * LROW];
    __shared__ __attribute__((aligned(16))) bf16_t ldsB[2][BN * LROW];

    const int mbase = blockIdx.x * BM, nb = blockIdx.y * BN;
    const int tid = threadIdx.x, wave = tid >> 5, lane = tid & 31;
    const int mh = wave >> 1, nw0 = (wave & 1) * 32;
    const float* asrc = emb + (size_t)mbase * DIM;

    v8f acc0 = {0.f,0.f,0.f,0.f,0.f,0.f,0.f,0.f};
    v8f acc1 = {0.f,0.f,0.f,0.f,0.f,0.f,0.f,0.f};

    stage_a_f32(asrc, DIM, 0, tid, ldsA[0]);
    stage_b_tile(sW1, NSHH, 0, nb, tid, ldsB[0]);
    const int nsteps = DIM / BK;
    for (int s = 0; s < nsteps; ++s) {
        __syncthreads();
        const int cur = s & 1;
        if (s + 1 < nsteps) {
            stage_a_f32(asrc, DIM, (s + 1) * BK, tid, ldsA[cur ^ 1]);
            stage_b_tile(sW1, NSHH, (s + 1) * BK, nb, tid, ldsB[cur ^ 1]);
        }
        v16bf a  = fragA(&ldsA[cur][mh * 16 * LROW], lane);
        v16bf b0 = fragB(&ldsB[cur][nw0 * LROW], lane);
        v16bf b1 = fragB(&ldsB[cur][(nw0 + 16) * LROW], lane);
        acc0 = wmma_bf16(a, b0, acc0);
        acc1 = wmma_bf16(a, b1, acc1);
    }

    const int half = lane >> 4;
#pragma unroll
    for (int sub = 0; sub < 2; ++sub) {
        v8f& acc = sub ? acc1 : acc0;
        int n = nb + nw0 + sub * 16 + (lane & 15);
        float bias = sB1[n];
#pragma unroll
        for (int r = 0; r < 8; ++r) {
            int m = mbase + mh * 16 + r + 8 * half;
            hs[(size_t)m * NSHH + n] = (bf16_t)silu_f(acc[r] + bias);
        }
    }
}

// ----- Kernel 4: shared MLP stage 2: out = hs @ sW2 + sB2 (fp32 base) -------
// grid (T/64, DIM/64); A tile DMA'd by the Tensor Data Mover

__global__ __launch_bounds__(256)
void moe_shared2_kernel(const bf16_t* __restrict__ hs, const float* __restrict__ sW2,
                        const float* __restrict__ sB2, float* __restrict__ out) {
    __shared__ __attribute__((aligned(16))) bf16_t ldsA[2][BM * LROW];
    __shared__ __attribute__((aligned(16))) bf16_t ldsB[2][BN * LROW];

    const int mbase = blockIdx.x * BM, nb = blockIdx.y * BN;
    const int tid = threadIdx.x, wave = tid >> 5, lane = tid & 31;
    const int mh = wave >> 1, nw0 = (wave & 1) * 32;
    const bf16_t* asrc = hs + (size_t)mbase * NSHH;
    const unsigned ldsA_addr[2] = {(unsigned)(uintptr_t)(void*)&ldsA[0][0],
                                   (unsigned)(uintptr_t)(void*)&ldsA[1][0]};

    v8f acc0 = {0.f,0.f,0.f,0.f,0.f,0.f,0.f,0.f};
    v8f acc1 = {0.f,0.f,0.f,0.f,0.f,0.f,0.f,0.f};

    if (wave == 0) tdm_load_tile64x32(ldsA_addr[0], asrc, NSHH);
    stage_b_tile(sW2, DIM, 0, nb, tid, ldsB[0]);
    const int nsteps = NSHH / BK;
    for (int s = 0; s < nsteps; ++s) {
        if (wave == 0) __builtin_amdgcn_s_wait_tensorcnt(0);
        __syncthreads();
        const int cur = s & 1;
        if (s + 1 < nsteps) {
            if (wave == 0)
                tdm_load_tile64x32(ldsA_addr[cur ^ 1], asrc + (s + 1) * BK, NSHH);
            stage_b_tile(sW2, DIM, (s + 1) * BK, nb, tid, ldsB[cur ^ 1]);
        }
        v16bf a  = fragA(&ldsA[cur][mh * 16 * LROW], lane);
        v16bf b0 = fragB(&ldsB[cur][nw0 * LROW], lane);
        v16bf b1 = fragB(&ldsB[cur][(nw0 + 16) * LROW], lane);
        acc0 = wmma_bf16(a, b0, acc0);
        acc1 = wmma_bf16(a, b1, acc1);
    }

    const int half = lane >> 4;
#pragma unroll
    for (int sub = 0; sub < 2; ++sub) {
        v8f& acc = sub ? acc1 : acc0;
        int n = nb + nw0 + sub * 16 + (lane & 15);
        float bias = sB2[n];
#pragma unroll
        for (int r = 0; r < 8; ++r) {
            int m = mbase + mh * 16 + r + 8 * half;
            out[(size_t)m * DIM + n] = acc[r] + bias;
        }
    }
}

// ----- Kernel 5: routed stage A: h = silu(a@W1+B1)*(a@W3+B3) (bf16) ---------
// grid (T/64, HDIM/64, NEXP); gathered rows, dual WMMA chain sharing A frag

__global__ __launch_bounds__(256)
void moe_expA_kernel(const float* __restrict__ emb, const float* __restrict__ x,
                     const float* __restrict__ W1, const float* __restrict__ B1,
                     const float* __restrict__ W3, const float* __restrict__ B3,
                     const int* __restrict__ counts, const int* __restrict__ offs,
                     const int* __restrict__ tokL, bf16_t* __restrict__ hR) {
    const int e = blockIdx.z;
    const int cnt = counts[e];
    const int mt = blockIdx.x * BM;
    if (mt >= cnt) return;

    __shared__ __attribute__((aligned(16))) bf16_t ldsA[2][BM * LROW];
    __shared__ __attribute__((aligned(16))) bf16_t ldsB1[2][BN * LROW];
    __shared__ __attribute__((aligned(16))) bf16_t ldsB3[2][BN * LROW];
    __shared__ int ldsTok[BM];

    const int nb = blockIdx.y * BN;
    const int tid = threadIdx.x, wave = tid >> 5, lane = tid & 31;
    const int mh = wave >> 1, nw0 = (wave & 1) * 32;
    const float* __restrict__ src = (e < NIN) ? x : emb;
    const float* __restrict__ w1p = W1 + (size_t)e * DIM * HDIM;
    const float* __restrict__ w3p = W3 + (size_t)e * DIM * HDIM;

    if (tid < BM) {
        int g = mt + tid;
        ldsTok[tid] = (g < cnt) ? tokL[e * T_TOK + g] : -1;
    }
    __syncthreads();

    auto stageAg = [&](int k0, bf16_t* dst) {
        int row = tid >> 2, c8 = (tid & 3) * 8;
        int tok = ldsTok[row];
        v8bf pk;
        if (tok >= 0) {
            const float* p = src + (size_t)tok * DIM + k0 + c8;
            float4 r0 = *(const float4*)(p);
            float4 r1 = *(const float4*)(p + 4);
            float a[8] = {r0.x, r0.y, r0.z, r0.w, r1.x, r1.y, r1.z, r1.w};
            for (int j = 0; j < 8; ++j) pk[j] = (bf16_t)a[j];
        } else {
            for (int j = 0; j < 8; ++j) pk[j] = (bf16_t)0.f;
        }
        *(v8bf*)(dst + row * LROW + c8) = pk;
    };

    v8f accG0 = {0.f,0.f,0.f,0.f,0.f,0.f,0.f,0.f};
    v8f accG1 = {0.f,0.f,0.f,0.f,0.f,0.f,0.f,0.f};
    v8f accU0 = {0.f,0.f,0.f,0.f,0.f,0.f,0.f,0.f};
    v8f accU1 = {0.f,0.f,0.f,0.f,0.f,0.f,0.f,0.f};

    stageAg(0, ldsA[0]);
    stage_b_tile(w1p, HDIM, 0, nb, tid, ldsB1[0]);
    stage_b_tile(w3p, HDIM, 0, nb, tid, ldsB3[0]);
    const int nsteps = DIM / BK;
    for (int s = 0; s < nsteps; ++s) {
        __syncthreads();
        const int cur = s & 1;
        if (s + 1 < nsteps) {
            stageAg((s + 1) * BK, ldsA[cur ^ 1]);
            stage_b_tile(w1p, HDIM, (s + 1) * BK, nb, tid, ldsB1[cur ^ 1]);
            stage_b_tile(w3p, HDIM, (s + 1) * BK, nb, tid, ldsB3[cur ^ 1]);
        }
        v16bf a   = fragA(&ldsA[cur][mh * 16 * LROW], lane);
        v16bf bg0 = fragB(&ldsB1[cur][nw0 * LROW], lane);
        v16bf bg1 = fragB(&ldsB1[cur][(nw0 + 16) * LROW], lane);
        v16bf bu0 = fragB(&ldsB3[cur][nw0 * LROW], lane);
        v16bf bu1 = fragB(&ldsB3[cur][(nw0 + 16) * LROW], lane);
        accG0 = wmma_bf16(a, bg0, accG0);
        accG1 = wmma_bf16(a, bg1, accG1);
        accU0 = wmma_bf16(a, bu0, accU0);
        accU1 = wmma_bf16(a, bu1, accU1);
    }

    const int half = lane >> 4;
    const int rowbase = offs[e];
#pragma unroll
    for (int sub = 0; sub < 2; ++sub) {
        v8f& aG = sub ? accG1 : accG0;
        v8f& aU = sub ? accU1 : accU0;
        int hcol = nb + nw0 + sub * 16 + (lane & 15);
        float b1s = B1[e * HDIM + hcol];
        float b3s = B3[e * HDIM + hcol];
#pragma unroll
        for (int r = 0; r < 8; ++r) {
            int g = mt + mh * 16 + r + 8 * half;
            if (g < cnt) {
                float gv = silu_f(aG[r] + b1s);
                float uv = aU[r] + b3s;
                hR[(size_t)(rowbase + g) * HDIM + hcol] = (bf16_t)(gv * uv);
            }
        }
    }
}

// ----- Kernel 6: routed stage B: out += w * (h @ W2 + B2) -------------------
// grid (T/64, DIM/64, NEXP); TDM-staged A, fp32 global atomics
// Rows past cnt DMA garbage, but each WMMA output row depends only on its own
// A row and those rows are masked in the epilogue.

__global__ __launch_bounds__(256)
void moe_expB_kernel(const bf16_t* __restrict__ hR, const float* __restrict__ W2,
                     const float* __restrict__ B2, const int* __restrict__ counts,
                     const int* __restrict__ offs, const int* __restrict__ tokL,
                     const float* __restrict__ wL, float* __restrict__ out) {
    const int e = blockIdx.z;
    const int cnt = counts[e];
    const int mt = blockIdx.x * BM;
    if (mt >= cnt) return;

    __shared__ __attribute__((aligned(16))) bf16_t ldsA[2][BM * LROW];
    __shared__ __attribute__((aligned(16))) bf16_t ldsB[2][BN * LROW];
    __shared__ int   ldsTok[BM];
    __shared__ float ldsW[BM];

    const int nb = blockIdx.y * BN;
    const int tid = threadIdx.x, wave = tid >> 5, lane = tid & 31;
    const int mh = wave >> 1, nw0 = (wave & 1) * 32;
    const float* __restrict__ w2p = W2 + (size_t)e * HDIM * DIM;
    const int rowbase = offs[e];
    const bf16_t* asrc = hR + (size_t)(rowbase + mt) * HDIM;
    const unsigned ldsA_addr[2] = {(unsigned)(uintptr_t)(void*)&ldsA[0][0],
                                   (unsigned)(uintptr_t)(void*)&ldsA[1][0]};

    if (tid < BM) {
        int g = mt + tid;
        ldsTok[tid] = (g < cnt) ? tokL[e * T_TOK + g] : -1;
        ldsW[tid]   = (g < cnt) ? wL[e * T_TOK + g] : 0.0f;
    }
    __syncthreads();

    v8f acc0 = {0.f,0.f,0.f,0.f,0.f,0.f,0.f,0.f};
    v8f acc1 = {0.f,0.f,0.f,0.f,0.f,0.f,0.f,0.f};

    if (wave == 0) tdm_load_tile64x32(ldsA_addr[0], asrc, HDIM);
    stage_b_tile(w2p, DIM, 0, nb, tid, ldsB[0]);
    const int nsteps = HDIM / BK;
    for (int s = 0; s < nsteps; ++s) {
        if (wave == 0) __builtin_amdgcn_s_wait_tensorcnt(0);
        __syncthreads();
        const int cur = s & 1;
        if (s + 1 < nsteps) {
            if (wave == 0)
                tdm_load_tile64x32(ldsA_addr[cur ^ 1], asrc + (s + 1) * BK, HDIM);
            stage_b_tile(w2p, DIM, (s + 1) * BK, nb, tid, ldsB[cur ^ 1]);
        }
        v16bf a  = fragA(&ldsA[cur][mh * 16 * LROW], lane);
        v16bf b0 = fragB(&ldsB[cur][nw0 * LROW], lane);
        v16bf b1 = fragB(&ldsB[cur][(nw0 + 16) * LROW], lane);
        acc0 = wmma_bf16(a, b0, acc0);
        acc1 = wmma_bf16(a, b1, acc1);
    }

    const int half = lane >> 4;
#pragma unroll
    for (int sub = 0; sub < 2; ++sub) {
        v8f& acc = sub ? acc1 : acc0;
        int col = nb + nw0 + sub * 16 + (lane & 15);
        float b2s = B2[e * DIM + col];
#pragma unroll
        for (int r = 0; r < 8; ++r) {
            int ml = mh * 16 + r + 8 * half;
            int g = mt + ml;
            if (g < cnt) {
                int tok = ldsTok[ml];
                float w = ldsW[ml];
                atomicAdd(&out[(size_t)tok * DIM + col], w * (acc[r] + b2s));
            }
        }
    }
}

// ---------------------------------------------------------------------------

extern "C" void kernel_launch(void* const* d_in, const int* in_sizes, int n_in,
                              void* d_out, int out_size, void* d_ws, size_t ws_size,
                              hipStream_t stream) {
    const float* emb = (const float*)d_in[0];
    const float* x   = (const float*)d_in[1];
    const float* gw  = (const float*)d_in[2];
    const float* W1  = (const float*)d_in[3];
    const float* B1  = (const float*)d_in[4];
    const float* W2  = (const float*)d_in[5];
    const float* B2  = (const float*)d_in[6];
    const float* W3  = (const float*)d_in[7];
    const float* B3  = (const float*)d_in[8];
    const float* sW1 = (const float*)d_in[9];
    const float* sB1 = (const float*)d_in[10];
    const float* sW2 = (const float*)d_in[11];
    const float* sB2 = (const float*)d_in[12];
    float* out = (float*)d_out;

    char* ws = (char*)d_ws;
    int*    counts = (int*)(ws + 0);
    int*    offs   = (int*)(ws + 64);
    int*    tokL   = (int*)(ws + 128);                    // E*T ints
    float*  wL     = (float*)(ws + 128 + 262144);         // E*T floats
    bf16_t* hR     = (bf16_t*)(ws + 524544);              // 2T*H bf16 = 16 MB
    bf16_t* hS     = (bf16_t*)(ws + 524544 + 16777216);   // T*2H bf16 = 16 MB

    moe_init_kernel<<<1, 32, 0, stream>>>(counts);
    moe_gate_kernel<<<T_TOK, 256, 0, stream>>>(emb, gw, counts, tokL, wL);
    moe_scan_kernel<<<1, 32, 0, stream>>>(counts, offs);

    moe_shared1_kernel<<<dim3(T_TOK / BM, NSHH / BN), 256, 0, stream>>>(emb, sW1, sB1, hS);
    moe_shared2_kernel<<<dim3(T_TOK / BM, DIM / BN), 256, 0, stream>>>(hS, sW2, sB2, out);

    moe_expA_kernel<<<dim3(T_TOK / BM, HDIM / BN, NEXP), 256, 0, stream>>>(
        emb, x, W1, B1, W3, B3, counts, offs, tokL, hR);
    moe_expB_kernel<<<dim3(T_TOK / BM, DIM / BN, NEXP), 256, 0, stream>>>(
        hR, W2, B2, counts, offs, tokL, wL, out);
}
